// MultiScaleGauRetention_22024592293953
// MI455X (gfx1250) — matrix operand order repaired and checked by reading, not compiled, not run on hardware
//
#include <hip/hip_runtime.h>
#include <cstdint>

// ---------- types ----------
typedef float  v8f   __attribute__((ext_vector_type(8)));
typedef __bf16 v16bf __attribute__((ext_vector_type(16)));
typedef __bf16 v2bf  __attribute__((ext_vector_type(2)));
typedef unsigned uvec4 __attribute__((ext_vector_type(4)));
typedef float  fvec4 __attribute__((ext_vector_type(4)));
typedef int    iv4   __attribute__((ext_vector_type(4)));
typedef int    iv8   __attribute__((ext_vector_type(8)));

// ---------- constants ----------
#define BB      2
#define TT      2048
#define HID     2048
#define QKD     1024
#define NHEADS  16
#define KD      64      // KEY_DIM
#define HD      256     // HEAD_DIM
#define EXPD    4096
#define NPROJ   10240   // 1024 q + 1024 k + 8192 h
#define SCALING 0.125f  // 64^-0.5

#if defined(__HIP_DEVICE_COMPILE__) && __has_builtin(__builtin_amdgcn_tensor_load_to_lds)
#define HAVE_TDM 1
#else
#define HAVE_TDM 0
#endif

// ---------- helpers ----------
__device__ __forceinline__ unsigned short f2bfraw(float f) {
  unsigned u = __builtin_bit_cast(unsigned, f);
  u += 0x7FFFu + ((u >> 16) & 1u);
  return (unsigned short)(u >> 16);
}
__device__ __forceinline__ __bf16 f2bf(float f) {
  unsigned short s = f2bfraw(f);
  return __builtin_bit_cast(__bf16, s);
}
__device__ __forceinline__ float bf2f(__bf16 h) {
  unsigned short s = __builtin_bit_cast(unsigned short, h);
  unsigned u = ((unsigned)s) << 16;
  return __builtin_bit_cast(float, u);
}
__device__ __forceinline__ v8f zero8() {
  v8f z;
#pragma unroll
  for (int i = 0; i < 8; ++i) z[i] = 0.f;
  return z;
}
__device__ __forceinline__ v8f wmma_bf16(v16bf a, v16bf b, v8f c) {
  return __builtin_amdgcn_wmma_f32_16x16x32_bf16(false, a, false, b, (short)0, c,
                                                 false, false);
}
struct bfpack { uvec4 lo, hi; };
__device__ __forceinline__ v16bf mkfrag(uvec4 lo, uvec4 hi) {
  bfpack p{lo, hi};
  return __builtin_bit_cast(v16bf, p);
}
struct wpack { unsigned w[8]; };
__device__ __forceinline__ unsigned ldsoff(const void* p) {
  return (unsigned)(unsigned long long)p;  // low 32 bits of generic ptr = LDS byte addr
}

#if HAVE_TDM
// 2D tile DMA: global (row-major, stride_elems bf16 per row) -> LDS, with
// pad_amount DWORDs inserted every pad_interval DWORDs (matches padded LDS rows).
// pad codes per CDNA5 ISA: interval 3 => 16 DW (32 bf16 row),
// 4 => 32 DW (64 bf16 row); amount 3 => 4 DW (8 bf16 pad).
__device__ __forceinline__ void tdm_load_2d(unsigned lds_off, const void* gptr,
                                            unsigned tile_x, unsigned tile_y,
                                            unsigned stride_elems,
                                            unsigned tensor_x, unsigned tensor_y,
                                            unsigned pad_int, unsigned pad_amt) {
  unsigned long long ga = (unsigned long long)gptr;
  uvec4 g0;
  g0[0] = 1u;                                   // count=1, user mode
  g0[1] = lds_off;                              // lds_addr
  g0[2] = (unsigned)ga;                         // global_addr[31:0]
  g0[3] = (unsigned)((ga >> 32) & 0x01FFFFFFu)  // global_addr[56:32]
          | (2u << 30);                         // type=2 (image)
  iv8 g1;
  g1[0] = (int)((1u << 16)                      // data_size = 2 bytes
                | (1u << 20)                    // pad_enable
                | (pad_int << 22) | (pad_amt << 25));
  g1[1] = (int)((tensor_x & 0xFFFFu) << 16);            // tensor_dim0[15:0]
  g1[2] = (int)((tensor_x >> 16) | ((tensor_y & 0xFFFFu) << 16));
  g1[3] = (int)((tensor_y >> 16) | (tile_x << 16));     // tile_dim0
  g1[4] = (int)tile_y;                                  // tile_dim1 (tile_dim2=0)
  g1[5] = (int)stride_elems;                            // tensor_dim0_stride lo
  g1[6] = 0;
  g1[7] = 0;
  iv4 z4 = {0, 0, 0, 0};
#if __clang_major__ >= 23
  iv8 z8 = {0, 0, 0, 0, 0, 0, 0, 0};
  __builtin_amdgcn_tensor_load_to_lds(g0, g1, z4, z4, z8, 0);
#else
  __builtin_amdgcn_tensor_load_to_lds(g0, g1, z4, z4, 0);
#endif
}
#endif

// ---------- kernel 1: fp32 -> bf16 convert ----------
__global__ void cvt_bf16_kernel(const float* __restrict__ src,
                                __bf16* __restrict__ dst, long n) {
  long i = (long)blockIdx.x * blockDim.x + threadIdx.x;
  if (i < n) dst[i] = f2bf(src[i]);
}

// ---------- kernel 2/6: bf16 GEMM  Y = (silu?)(X @ W^T) ----------
// Block tile 128x64, 256 threads (8 waves), each wave 32x32 via 2x2 WMMA frags.
template <bool SILU, bool OUTBF>
__global__ __launch_bounds__(256) void gemm_bf16_nt(
    const __bf16* __restrict__ X, const __bf16* __restrict__ W,
    float* __restrict__ Yf, __bf16* __restrict__ Yh, int M, int N, int K,
    int sclBeg, int sclEnd, float scl) {
  __shared__ __bf16 lx[128][40];
  __shared__ __bf16 lw[64][40];
  const int tid = threadIdx.x;
  const int lane = tid & 31, wv = tid >> 5;
  const int hi = lane >> 4, ln = lane & 15;
  const int m0 = blockIdx.y * 128, n0 = blockIdx.x * 64;
  const int mo = (wv >> 1) * 32, no = (wv & 1) * 32;

  v8f c[2][2];
#pragma unroll
  for (int i = 0; i < 2; ++i)
#pragma unroll
    for (int j = 0; j < 2; ++j) c[i][j] = zero8();

  const int xr = tid >> 2, xs = tid & 3;
  (void)xs;
  for (int k0 = 0; k0 < K; k0 += 32) {
    __syncthreads();
#if HAVE_TDM
    if (wv == 0) {
      tdm_load_2d(ldsoff(&lx[0][0]), &X[(size_t)m0 * K + k0], 32, 128, K,
                  (unsigned)K, (unsigned)M, 3, 3);
      tdm_load_2d(ldsoff(&lw[0][0]), &W[(size_t)n0 * K + k0], 32, 64, K,
                  (unsigned)K, (unsigned)N, 3, 3);
      __builtin_amdgcn_s_wait_tensorcnt(0);
    }
#else
#pragma unroll
    for (int it = 0; it < 2; ++it) {
      int idx = tid + it * 256;
      int row = idx >> 2, seg = idx & 3;
      *(uvec4*)&lx[row][seg * 8] =
          *(const uvec4*)&X[(size_t)(m0 + row) * K + k0 + seg * 8];
    }
    {
      int row = tid >> 2, seg = tid & 3;
      *(uvec4*)&lw[row][seg * 8] =
          *(const uvec4*)&W[(size_t)(n0 + row) * K + k0 + seg * 8];
    }
#endif
    if (k0 + 32 < K)
      __builtin_prefetch(&X[(size_t)(m0 + xr) * K + k0 + 32], 0, 1);
    __syncthreads();

    v16bf a[2], bb[2];
#pragma unroll
    for (int mi = 0; mi < 2; ++mi) {
      const __bf16* r = &lx[mo + mi * 16 + ln][0];
      // A layout: elems 0..7 -> K = 8*hi + e ; elems 8..15 -> K = 16 + 8*hi + e
      a[mi] = mkfrag(*(const uvec4*)&r[8 * hi], *(const uvec4*)&r[16 + 8 * hi]);
    }
#pragma unroll
    for (int ni = 0; ni < 2; ++ni) {
      const __bf16* r = &lw[no + ni * 16 + ln][0];
      // B layout: elems 0..7 -> K = 8*hi + e ; elems 8..15 -> K = 16 + 8*hi + e
      bb[ni] = mkfrag(*(const uvec4*)&r[8 * hi], *(const uvec4*)&r[16 + 8 * hi]);
    }
#pragma unroll
    for (int mi = 0; mi < 2; ++mi)
#pragma unroll
      for (int ni = 0; ni < 2; ++ni)
        c[mi][ni] = wmma_bf16(a[mi], bb[ni], c[mi][ni]);
  }

  // epilogue: C frag lane: N = ln, element r: M = r + 8*hi
#pragma unroll
  for (int mi = 0; mi < 2; ++mi)
#pragma unroll
    for (int ni = 0; ni < 2; ++ni)
#pragma unroll
      for (int r = 0; r < 8; ++r) {
        int row = m0 + mo + mi * 16 + r + 8 * hi;
        int col = n0 + no + ni * 16 + ln;
        float v = c[mi][ni][r];
        if (SILU) v = v / (1.0f + __expf(-v));
        if (col >= sclBeg && col < sclEnd) v *= scl;
        if (OUTBF) Yh[(size_t)row * N + col] = f2bf(v);
        else       Yf[(size_t)row * N + col] = v;
      }
}

// ---------- kernel 3: rotary theta-shift on q,k -> head-major bf16 ----------
__global__ void rotary_kernel(const __bf16* __restrict__ yproj,
                              const float* __restrict__ sinp,
                              const float* __restrict__ cosp,
                              __bf16* __restrict__ qrot,
                              __bf16* __restrict__ krot) {
  long idx = (long)blockIdx.x * 256 + threadIdx.x;  // [0, B*T*H*32)
  int pr = (int)(idx & 31);
  long r1 = idx >> 5;
  int h = (int)(r1 & 15);
  long r2 = r1 >> 4;
  int t = (int)(r2 & (TT - 1));
  int b = (int)(r2 >> 11);
  int d0 = pr * 2;
  const __bf16* row = yproj + ((size_t)(b * TT + t)) * NPROJ;
  float q0 = bf2f(row[h * KD + d0]), q1 = bf2f(row[h * KD + d0 + 1]);
  float k0 = bf2f(row[QKD + h * KD + d0]), k1 = bf2f(row[QKD + h * KD + d0 + 1]);
  float s0 = sinp[t * KD + d0], s1 = sinp[t * KD + d0 + 1];
  float c0 = cosp[t * KD + d0], c1 = cosp[t * KD + d0 + 1];
  size_t base = (((size_t)(b * NHEADS + h)) * TT + t) * KD + d0;
  qrot[base + 0] = f2bf(q0 * c0 - q1 * s0);
  qrot[base + 1] = f2bf(q1 * c1 + q0 * s1);
  krot[base + 0] = f2bf(k0 * c0 - k1 * s0);
  krot[base + 1] = f2bf(k1 * c1 + k0 * s1);
}

// ---------- kernel 4: retention core ----------
// grid: (T/64, B*H), block 128 (4 waves). Each wave owns 16 t-rows.
// c = k * q^T  (C: M=s, N=t) so the C fragment IS the A fragment for qk@v.
__global__ __launch_bounds__(128) void retention_kernel(
    const __bf16* __restrict__ qrot, const __bf16* __restrict__ krot,
    const __bf16* __restrict__ yproj, const float* __restrict__ decay,
    float* __restrict__ attn) {
  __shared__ __bf16 lq[64][72];    // q block, padded rows (144 B)
  __shared__ __bf16 lk[32][72];    // k tile
  __shared__ __bf16 lvT[256][40];  // v tile TRANSPOSED: [vcol][s], 80 B rows
  __shared__ float lden[4][16];

  const int tid = threadIdx.x;
  const int lane = tid & 31, wv = tid >> 5;
  const int hi = lane >> 4, ln = lane & 15;
  const int bh = blockIdx.y;
  const int b = bh >> 4, h = bh & 15;
  const int T0 = blockIdx.x * 64;

  // stage q block (64 x 64)
  const __bf16* qbase = qrot + ((size_t)bh * TT + T0) * KD;
#if HAVE_TDM
  if (wv == 0) {
    tdm_load_2d(ldsoff(&lq[0][0]), qbase, 64, 64, KD, KD,
                (unsigned)(BB * NHEADS * TT), 4, 3);
    __builtin_amdgcn_s_wait_tensorcnt(0);
  }
#else
#pragma unroll
  for (int it = 0; it < 4; ++it) {
    int idx = tid + it * 128;
    int row = idx >> 3, seg = idx & 7;
    *(uvec4*)&lq[row][seg * 8] = *(const uvec4*)&qbase[(size_t)row * KD + seg * 8];
  }
#endif

  v8f acc[16];
#pragma unroll
  for (int i = 0; i < 16; ++i) acc[i] = zero8();
  float den = 0.f;

  const int tloc = wv * 16 + ln;  // lane's t column (local)
  const float* drow = decay + ((size_t)h * TT + (T0 + tloc)) * TT;

  const int sblocks = (T0 + 64) / 32;
  for (int sb = 0; sb < sblocks; ++sb) {
    const int s0 = sb * 32;
    __syncthreads();
    // stage k tile (32 x 64)
#if HAVE_TDM
    if (wv == 0) {
      tdm_load_2d(ldsoff(&lk[0][0]), &krot[((size_t)bh * TT + s0) * KD], 64, 32,
                  KD, KD, (unsigned)(BB * NHEADS * TT), 4, 3);
      __builtin_amdgcn_s_wait_tensorcnt(0);
    }
#else
#pragma unroll
    for (int it = 0; it < 2; ++it) {
      int idx = tid + it * 128;
      int row = idx >> 3, seg = idx & 7;
      *(uvec4*)&lk[row][seg * 8] =
          *(const uvec4*)&krot[((size_t)bh * TT + s0 + row) * KD + seg * 8];
    }
#endif
    // stage v tile (32 x 256) TRANSPOSED into lvT[col][s]
#pragma unroll
    for (int it = 0; it < 8; ++it) {
      int idx = tid + it * 128;
      int row = idx >> 5, seg = idx & 31;  // row = s-local, cols seg*8..+7
      uvec4 d = *(const uvec4*)&yproj[((size_t)(b * TT + s0 + row)) * NPROJ +
                                      2 * QKD + (size_t)h * HD + seg * 8];
#pragma unroll
      for (int j = 0; j < 4; ++j) {
        v2bf pr = __builtin_bit_cast(v2bf, d[j]);
        lvT[seg * 8 + 2 * j + 0][row] = pr[0];
        lvT[seg * 8 + 2 * j + 1][row] = pr[1];
      }
    }
    __syncthreads();

    // qk^T: A = k rows (M=s), B = q^T (N=t)
    v8f c0 = zero8(), c1 = zero8();
#pragma unroll
    for (int ks = 0; ks < 2; ++ks) {
      const __bf16* kr0 = &lk[ln][ks * 32];
      const __bf16* kr1 = &lk[16 + ln][ks * 32];
      const __bf16* qr  = &lq[tloc][ks * 32];
      v16bf a0 = mkfrag(*(const uvec4*)&kr0[8 * hi], *(const uvec4*)&kr0[16 + 8 * hi]);
      v16bf a1 = mkfrag(*(const uvec4*)&kr1[8 * hi], *(const uvec4*)&kr1[16 + 8 * hi]);
      v16bf bq = mkfrag(*(const uvec4*)&qr[8 * hi], *(const uvec4*)&qr[16 + 8 * hi]);
      c0 = wmma_bf16(a0, bq, c0);
      c1 = wmma_bf16(a1, bq, c1);
    }

    // decay multiply + denominator accumulation (lane t = tloc, elem s)
    fvec4 d0a = *(const fvec4*)&drow[s0 + 8 * hi];
    fvec4 d0b = *(const fvec4*)&drow[s0 + 8 * hi + 4];
    fvec4 d1a = *(const fvec4*)&drow[s0 + 16 + 8 * hi];
    fvec4 d1b = *(const fvec4*)&drow[s0 + 16 + 8 * hi + 4];
#pragma unroll
    for (int r = 0; r < 4; ++r) {
      c0[r] *= d0a[r]; c0[r + 4] *= d0b[r];
      c1[r] *= d1a[r]; c1[r + 4] *= d1b[r];
      den += fabsf(c0[r]) + fabsf(c0[r + 4]) + fabsf(c1[r]) + fabsf(c1[r + 4]);
    }

    // C(s,t) frag == A(t,K=s) frag: pack to bf16 dwords, single bit_cast
    wpack aw;
#pragma unroll
    for (int i = 0; i < 4; ++i) {
      aw.w[i]     = (unsigned)f2bfraw(c0[2 * i]) | ((unsigned)f2bfraw(c0[2 * i + 1]) << 16);
      aw.w[4 + i] = (unsigned)f2bfraw(c1[2 * i]) | ((unsigned)f2bfraw(c1[2 * i + 1]) << 16);
    }
    v16bf af = __builtin_bit_cast(v16bf, aw);

    // out += qk @ v over 16 column tiles; B frag = two 8-wide s-runs in lvT
#pragma unroll
    for (int ni = 0; ni < 16; ++ni) {
      const __bf16* vp = &lvT[ni * 16 + ln][0];  // K = 8*hi + e ; 16 + 8*hi + e
      v16bf bv = mkfrag(*(const uvec4*)&vp[8 * hi], *(const uvec4*)&vp[16 + 8 * hi]);
      acc[ni] = wmma_bf16(af, bv, acc[ni]);
    }
  }

  // reduce denominator across the two half-wave s-subsets (same t both halves)
  den += __shfl_xor(den, 16, 32);
  if (hi == 0) lden[wv][ln] = fminf(fmaxf(den, 1.0f), 50000.0f);
  __syncthreads();

  // write: acc lane N(vcol)=ln, element r: M(t)=r+8*hi
  float* obase = attn + ((size_t)(b * TT + T0 + wv * 16)) * EXPD + (size_t)h * HD;
#pragma unroll
  for (int ni = 0; ni < 16; ++ni)
#pragma unroll
    for (int r = 0; r < 8; ++r) {
      int trow = r + 8 * hi;
      obase[(size_t)trow * EXPD + ni * 16 + ln] = acc[ni][r] / lden[wv][trow];
    }
}

// ---------- kernel 5: RMS-norm over HEAD_DIM, * gn_w * gate, -> bf16 ----------
__global__ __launch_bounds__(256) void rms_gate_kernel(
    const float* __restrict__ attn, const __bf16* __restrict__ yproj,
    const float* __restrict__ gnw, __bf16* __restrict__ aob) {
  const int bi = blockIdx.x;
  const int h = bi & 15;
  const int tl = bi >> 4;  // b*T + t
  const int i = threadIdx.x;
  const float* base = attn + (size_t)tl * EXPD + (size_t)h * HD;
  float x = base[i];
  float p = x * x;
#pragma unroll
  for (int m = 1; m < 32; m <<= 1) p += __shfl_xor(p, m, 32);
  __shared__ float red[8];
  if ((i & 31) == 0) red[i >> 5] = p;
  __syncthreads();
  float s = 0.f;
#pragma unroll
  for (int w = 0; w < 8; ++w) s += red[w];
  float rinv = rsqrtf(s * (1.0f / 256.0f) + 1e-6f);
  float gate = bf2f(yproj[(size_t)tl * NPROJ + 2 * QKD + EXPD + (size_t)h * HD + i]);
  aob[(size_t)tl * EXPD + (size_t)h * HD + i] = f2bf(gnw[i] * x * rinv * gate);
}

// ---------- host ----------
extern "C" void kernel_launch(void* const* d_in, const int* in_sizes, int n_in,
                              void* d_out, int out_size, void* d_ws, size_t ws_size,
                              hipStream_t stream) {
  (void)in_sizes; (void)n_in; (void)out_size; (void)ws_size;
  const float* x     = (const float*)d_in[0];
  const float* sinp  = (const float*)d_in[1];
  const float* cosp  = (const float*)d_in[2];
  const float* decay = (const float*)d_in[3];
  const float* Wq    = (const float*)d_in[4];
  const float* Wk    = (const float*)d_in[5];
  const float* Wh    = (const float*)d_in[6];
  const float* Wo    = (const float*)d_in[7];
  const float* gnw   = (const float*)d_in[8];
  float* out = (float*)d_out;

  char* ws = (char*)d_ws;
  size_t off = 0;
  __bf16* xb    = (__bf16*)(ws + off); off += (size_t)BB * TT * HID * 2;
  __bf16* wqkh  = (__bf16*)(ws + off); off += (size_t)NPROJ * HID * 2;
  __bf16* wob   = (__bf16*)(ws + off); off += (size_t)HID * EXPD * 2;
  __bf16* yproj = (__bf16*)(ws + off); off += (size_t)BB * TT * NPROJ * 2;
  __bf16* qrot  = (__bf16*)(ws + off); off += (size_t)BB * NHEADS * TT * KD * 2;
  __bf16* krot  = (__bf16*)(ws + off); off += (size_t)BB * NHEADS * TT * KD * 2;
  float*  attn  = (float*)(ws + off);  off += (size_t)BB * TT * EXPD * 4;
  __bf16* aob   = (__bf16*)(ws + off); off += (size_t)BB * TT * EXPD * 2;

  // 1) convert to bf16
  {
    long n;
    n = (long)BB * TT * HID;
    cvt_bf16_kernel<<<(n + 255) / 256, 256, 0, stream>>>(x, xb, n);
    n = (long)QKD * HID;
    cvt_bf16_kernel<<<(n + 255) / 256, 256, 0, stream>>>(Wq, wqkh, n);
    cvt_bf16_kernel<<<(n + 255) / 256, 256, 0, stream>>>(Wk, wqkh + (size_t)QKD * HID, n);
    n = (long)2 * EXPD * HID;
    cvt_bf16_kernel<<<(n + 255) / 256, 256, 0, stream>>>(Wh, wqkh + (size_t)2 * QKD * HID, n);
    n = (long)HID * EXPD;
    cvt_bf16_kernel<<<(n + 255) / 256, 256, 0, stream>>>(Wo, wob, n);
  }

  // 2) fused projection: yproj = silu(x @ [Wq;Wk;Wh]^T), k cols scaled
  {
    dim3 g(NPROJ / 64, (BB * TT) / 128);
    gemm_bf16_nt<true, true><<<g, 256, 0, stream>>>(
        xb, wqkh, nullptr, yproj, BB * TT, NPROJ, HID, QKD, 2 * QKD, SCALING);
  }

  // 3) rotary
  {
    long n = (long)BB * TT * NHEADS * (KD / 2);
    rotary_kernel<<<n / 256, 256, 0, stream>>>(yproj, sinp, cosp, qrot, krot);
  }

  // 4) retention
  {
    dim3 g(TT / 64, BB * NHEADS);
    retention_kernel<<<g, 128, 0, stream>>>(qrot, krot, yproj, decay, attn);
  }

  // 5) rms norm * gate
  rms_gate_kernel<<<BB * TT * NHEADS, 256, 0, stream>>>(attn, yproj, gnw, aob);

  // 6) output projection: out = aob @ Wo^T (fp32 out)
  {
    dim3 g(HID / 64, (BB * TT) / 128);
    gemm_bf16_nt<false, false><<<g, 256, 0, stream>>>(
        aob, wob, out, nullptr, BB * TT, HID, EXPD, 0, 0, 1.0f);
  }
}
